// AMEODE_90159953478503
// MI455X (gfx1250) — matrix operand order repaired and checked by reading, not compiled, not run on hardware
//
#include <hip/hip_runtime.h>

typedef __attribute__((ext_vector_type(16))) _Float16 v16h;
typedef __attribute__((ext_vector_type(8)))  _Float16 v8h;
typedef __attribute__((ext_vector_type(8)))  float    v8f;

namespace {
constexpr int D = 32, E = 8, T = 10;
constexpr int WAVES = 4, ROWS = 16, BLOCK = WAVES * 32, ROWS_PER_BLOCK = WAVES * ROWS;

// f16 LDS pitches (halves) - all multiples of 8 (16B aligned rows)
constexpr int P64 = 72;   // 64-wide weight rows (+8 pad)
constexpr int P32 = 40;   // 32-wide rows (+8 pad)
constexpr int PG2 = 24;   // gate-2 (16-wide padded) rows
constexpr int PXIN = 40, PACT = 72;
constexpr int PFF = 36;   // f32 expert-dynamics rows (32 + 4 pad), conflict-free

// f16 region offsets (in halves)
constexpr int OW1 = 0;
constexpr int OW2 = OW1 + E * 32 * P64;   // 18432
constexpr int OW3 = OW2 + E * 64 * P64;   // 55296
constexpr int OG1 = OW3 + E * 64 * P32;   // 75776
constexpr int OG2 = OG1 + 64 * P64;       // 80384
constexpr int OWV = OG2 + 64 * PG2;       // 81920
constexpr int WXIN = 0;
constexpr int WACT = WXIN + 16 * PXIN;    // 640
constexpr int WDX  = WACT + 16 * PACT;    // 1792
constexpr int WAVE_H = WDX + 16 * PXIN;   // 2432
constexpr int H_TOTAL = OWV + WAVES * WAVE_H;   // 91648

// f32 region offsets (in floats)
constexpr int FB1 = 0;
constexpr int FW1T = FB1 + E * 64;
constexpr int FB2 = FW1T + E * 64;
constexpr int FB3 = FB2 + E * 64;
constexpr int FGB1 = FB3 + E * 32;
constexpr int FGB2 = FGB1 + 64;
constexpr int FWAVE = FGB2 + 16;                // 1872
constexpr int WAVE_F = 16 * 8 + E * 16 * PFF;   // wgt + f32 expert dynamics = 4736
constexpr int F_TOTAL = FWAVE + WAVES * WAVE_F; // 20816
constexpr size_t SMEM_BYTES = (size_t)H_TOTAL * 2 + (size_t)F_TOTAL * 4; // 266560
} // namespace

__device__ __forceinline__ v8f vfill(float x) {
  v8f r;
#pragma unroll
  for (int i = 0; i < 8; ++i) r[i] = x;
  return r;
}

// tanh(x + b) with pre-doubled bias b2 = 2b:  1 - 2/(exp(2x+b2)+1)
__device__ __forceinline__ float tanh_b(float x, float b2) {
  float e = __expf(__builtin_fmaf(2.0f, x, b2));
  return 1.0f - 2.0f * __builtin_amdgcn_rcpf(e + 1.0f);
}

union HF16 { v16h v; v8h p[2]; };

// A fragment (16x32, f16): lane l holds row m=l&15; halves j<8 -> k=8g+j,
// j>=8 -> k=16+8g+(j-8), g=l>>4. Two contiguous 16B chunks per lane.
__device__ __forceinline__ v16h load_a32(const _Float16* row_base, int pitch, int lane) {
  int m = lane & 15, g = lane >> 4;
  HF16 a;
  const _Float16* p = row_base + m * pitch + 8 * g;
  a.p[0] = *(const v8h*)p;
  a.p[1] = *(const v8h*)(p + 16);
  return a.v;
}

// B fragment (32x16, f16): lane l holds col n=n0+(l&15); halves j<8 -> k=8g+j,
// j>=8 -> k=16+8g+(j-8), g=l>>4.
__device__ __forceinline__ v16h load_b32(const _Float16* kblock, int pitch, int lane, int n0) {
  v16h b;
  int g = lane >> 4;
  const _Float16* p = kblock + n0 + (lane & 15);
#pragma unroll
  for (int j = 0; j < 8; ++j) {
    b[j] = p[(8 * g + j) * pitch];
    b[8 + j] = p[(16 + 8 * g + j) * pitch];
  }
  return b;
}

__device__ __forceinline__ v8f wmma_f16(v16h a, v16h b, v8f c) {
  return __builtin_amdgcn_wmma_f32_16x16x32_f16(false, a, false, b, (short)0, c, false, false);
}

__global__ __launch_bounds__(BLOCK)
void AMEODE_90159953478503_kernel(
    const float* __restrict__ x0, const float* __restrict__ t_span,
    const float* __restrict__ W1, const float* __restrict__ b1,
    const float* __restrict__ W2, const float* __restrict__ b2,
    const float* __restrict__ W3, const float* __restrict__ b3,
    const float* __restrict__ Gw1, const float* __restrict__ Gb1,
    const float* __restrict__ Gw2, const float* __restrict__ Gb2,
    float* __restrict__ out)
{
  extern __shared__ char smem[];
  _Float16* hb = (_Float16*)smem;
  float* fb = (float*)(smem + (size_t)H_TOTAL * 2);
  const int tid = threadIdx.x;

  // ---- stage weights f32 -> f16 into LDS (once per block) ----
  for (int i = tid; i < E * 32 * 64; i += BLOCK) {
    int e = i >> 11, r = (i >> 6) & 31, n = i & 63;
    hb[OW1 + (e * 32 + r) * P64 + n] = (_Float16)W1[(e * 33 + r) * 64 + n];
  }
  for (int i = tid; i < E * 64; i += BLOCK) {
    int e = i >> 6, n = i & 63;
    fb[FW1T + i] = W1[(e * 33 + 32) * 64 + n];  // t-row of W1 -> dynamic bias
    fb[FB1 + i] = b1[i];
    fb[FB2 + i] = b2[i];
  }
  for (int i = tid; i < E * 64 * 64; i += BLOCK) {
    int e = i >> 12, r = (i >> 6) & 63, n = i & 63;
    hb[OW2 + (e * 64 + r) * P64 + n] = (_Float16)W2[i];
  }
  for (int i = tid; i < E * 64 * 32; i += BLOCK) {
    int e = i >> 11, r = (i >> 5) & 63, n = i & 31;
    hb[OW3 + (e * 64 + r) * P32 + n] = (_Float16)W3[i];
  }
  for (int i = tid; i < E * 32; i += BLOCK) fb[FB3 + i] = b3[i];
  for (int i = tid; i < 64 * 64; i += BLOCK) {
    int r = i >> 6, n = i & 63;
    hb[OG1 + r * P64 + n] = (_Float16)Gw1[i];
  }
  for (int i = tid; i < 64; i += BLOCK) fb[FGB1 + i] = Gb1[i];
  for (int i = tid; i < 64 * 16; i += BLOCK) {
    int r = i >> 4, n = i & 15;
    hb[OG2 + r * PG2 + n] = (n < 8) ? (_Float16)Gw2[r * 8 + n] : (_Float16)0.0f;
  }
  for (int i = tid; i < 16; i += BLOCK) fb[FGB2 + i] = (i < 8) ? Gb2[i] : 0.0f;
  __syncthreads();

  const int wave = tid >> 5, lane = tid & 31;
  const int hi = lane >> 4, nl = lane & 15;
  _Float16* xin = hb + OWV + wave * WAVE_H + WXIN;
  _Float16* act = hb + OWV + wave * WAVE_H + WACT;
  _Float16* dxb = hb + OWV + wave * WAVE_H + WDX;
  float* wgt  = fb + FWAVE + wave * WAVE_F;
  float* fstf = wgt + 16 * 8;

  const long r0 = (long)(blockIdx.x * WAVES + wave) * ROWS;

  // load x state tile (C/D layout: lane l -> col n=l&15 (+16*nt), rows m=v+8*(l>>4))
  v8f x[2];
#pragma unroll
  for (int nt = 0; nt < 2; ++nt)
#pragma unroll
    for (int v = 0; v < 8; ++v)
      x[nt][v] = x0[(r0 + v + 8 * hi) * D + nt * 16 + nl];

  // trajectory t=0
#pragma unroll
  for (int nt = 0; nt < 2; ++nt)
#pragma unroll
    for (int v = 0; v < 8; ++v)
      out[((r0 + v + 8 * hi) * T + 0) * D + nt * 16 + nl] = x[nt][v];

  auto store_xin = [&](const v8f* ks, float c) {
#pragma unroll
    for (int nt = 0; nt < 2; ++nt)
#pragma unroll
      for (int v = 0; v < 8; ++v)
        xin[(v + 8 * hi) * PXIN + nt * 16 + nl] = (_Float16)(x[nt][v] + c * ks[nt][v]);
  };

  auto dyn = [&](float t, v8f* kout) {
    const v8f zc = {};                         // inline-0 C operand for WMMA
    v16h ax = load_a32(xin, PXIN, lane);
    v8f fsum[2] = { vfill(0.f), vfill(0.f) };
#pragma unroll 1
    for (int e = 0; e < E; ++e) {
      const _Float16* w1e = hb + OW1 + e * 32 * P64;
      const _Float16* w2e = hb + OW2 + e * 64 * P64;
      const _Float16* w3e = hb + OW3 + e * 64 * P32;
      v8f h[4];
      // layer 1: x[16x32] @ W1[32x64]; bias (b1 + t*W1_trow) folded into tanh
#pragma unroll
      for (int nt = 0; nt < 4; ++nt)
        h[nt] = wmma_f16(ax, load_b32(w1e, P64, lane, nt * 16), zc);
#pragma unroll
      for (int nt = 0; nt < 4; ++nt) {
        float b2v = 2.0f * __builtin_fmaf(t, fb[FW1T + e * 64 + nt * 16 + nl],
                                          fb[FB1 + e * 64 + nt * 16 + nl]);
#pragma unroll
        for (int v = 0; v < 8; ++v)
          act[(v + 8 * hi) * PACT + nt * 16 + nl] = (_Float16)tanh_b(h[nt][v], b2v);
      }
      // layer 2: h1[16x64] @ W2[64x64]
      v16h a0 = load_a32(act, PACT, lane);
      v16h a1 = load_a32(act + 32, PACT, lane);
#pragma unroll
      for (int nt = 0; nt < 4; ++nt) {
        v8f acc = wmma_f16(a0, load_b32(w2e, P64, lane, nt * 16), zc);
        acc = wmma_f16(a1, load_b32(w2e + 32 * P64, P64, lane, nt * 16), acc);
        h[nt] = acc;
      }
#pragma unroll
      for (int nt = 0; nt < 4; ++nt) {
        float b2v = 2.0f * fb[FB2 + e * 64 + nt * 16 + nl];
#pragma unroll
        for (int v = 0; v < 8; ++v)
          act[(v + 8 * hi) * PACT + nt * 16 + nl] = (_Float16)tanh_b(h[nt][v], b2v);
      }
      // layer 3: h2[16x64] @ W3[64x32] -> f_e (f32 in LDS)
      a0 = load_a32(act, PACT, lane);
      a1 = load_a32(act + 32, PACT, lane);
#pragma unroll
      for (int nt = 0; nt < 2; ++nt) {
        v8f acc = wmma_f16(a0, load_b32(w3e, P32, lane, nt * 16), zc);
        acc = wmma_f16(a1, load_b32(w3e + 32 * P32, P32, lane, nt * 16), acc);
        float bv = fb[FB3 + e * 32 + nt * 16 + nl];
#pragma unroll
        for (int v = 0; v < 8; ++v) {
          float fv = acc[v] + bv;
          fsum[nt][v] += fv;
          fstf[(e * 16 + v + 8 * hi) * PFF + nt * 16 + nl] = fv;
        }
      }
    }
    // dx_init = mean over experts
#pragma unroll
    for (int nt = 0; nt < 2; ++nt)
#pragma unroll
      for (int v = 0; v < 8; ++v)
        dxb[(v + 8 * hi) * PXIN + nt * 16 + nl] = (_Float16)(fsum[nt][v] * 0.125f);
    // gate layer 1: [x, dx][16x64] @ Gw1[64x64]
    v16h adx = load_a32(dxb, PXIN, lane);
    v8f g[4];
#pragma unroll
    for (int nt = 0; nt < 4; ++nt) {
      v8f acc = wmma_f16(ax,  load_b32(hb + OG1, P64, lane, nt * 16), zc);
      acc = wmma_f16(adx, load_b32(hb + OG1 + 32 * P64, P64, lane, nt * 16), acc);
      g[nt] = acc;
    }
#pragma unroll
    for (int nt = 0; nt < 4; ++nt) {
      float b2v = 2.0f * fb[FGB1 + nt * 16 + nl];
#pragma unroll
      for (int v = 0; v < 8; ++v)
        act[(v + 8 * hi) * PACT + nt * 16 + nl] = (_Float16)tanh_b(g[nt][v], b2v);
    }
    // gate layer 2: g[16x64] @ Gw2pad[64x16] (cols 8..15 zero)
    v16h ag0 = load_a32(act, PACT, lane);
    v16h ag1 = load_a32(act + 32, PACT, lane);
    v8f lg = wmma_f16(ag0, load_b32(hb + OG2, PG2, lane, 0), zc);
    lg = wmma_f16(ag1, load_b32(hb + OG2 + 32 * PG2, PG2, lane, 0), lg);
    if (nl < 8) {
      float gb2v = fb[FGB2 + nl];
#pragma unroll
      for (int v = 0; v < 8; ++v) wgt[(v + 8 * hi) * 8 + nl] = lg[v] + gb2v;
    }
    // softmax over 8 experts per row (lanes 0..15 each own one row)
    if (lane < 16) {
      float vals[8]; float mx = -3.0e38f;
#pragma unroll
      for (int e = 0; e < 8; ++e) { vals[e] = wgt[lane * 8 + e]; mx = fmaxf(mx, vals[e]); }
      float s = 0.f;
#pragma unroll
      for (int e = 0; e < 8; ++e) { vals[e] = __expf(vals[e] - mx); s += vals[e]; }
      float r = __builtin_amdgcn_rcpf(s);
#pragma unroll
      for (int e = 0; e < 8; ++e) wgt[lane * 8 + e] = vals[e] * r;
    }
    // weighted sum of expert dynamics (all f32, no conversions)
    kout[0] = vfill(0.f); kout[1] = vfill(0.f);
#pragma unroll 1
    for (int e = 0; e < E; ++e) {
      float wv[8];
#pragma unroll
      for (int v = 0; v < 8; ++v) wv[v] = wgt[(v + 8 * hi) * 8 + e];
#pragma unroll
      for (int nt = 0; nt < 2; ++nt)
#pragma unroll
        for (int v = 0; v < 8; ++v)
          kout[nt][v] += wv[v] * fstf[(e * 16 + v + 8 * hi) * PFF + nt * 16 + nl];
    }
  };

  // ---- RK4 time loop (per-wave, fully on-chip) ----
#pragma unroll 1
  for (int s = 0; s < T - 1; ++s) {
    float t0 = t_span[s], t1 = t_span[s + 1];
    float dt = t1 - t0;
    v8f k[2], acc[2];
    store_xin(x, 0.0f);
    dyn(t0, k);
    acc[0] = k[0]; acc[1] = k[1];
    store_xin(k, 0.5f * dt);
    dyn(t0 + 0.5f * dt, k);
#pragma unroll
    for (int nt = 0; nt < 2; ++nt)
#pragma unroll
      for (int v = 0; v < 8; ++v) acc[nt][v] += 2.0f * k[nt][v];
    store_xin(k, 0.5f * dt);
    dyn(t0 + 0.5f * dt, k);
#pragma unroll
    for (int nt = 0; nt < 2; ++nt)
#pragma unroll
      for (int v = 0; v < 8; ++v) acc[nt][v] += 2.0f * k[nt][v];
    store_xin(k, dt);
    dyn(t1, k);
    float c6 = dt * (1.0f / 6.0f);
#pragma unroll
    for (int nt = 0; nt < 2; ++nt)
#pragma unroll
      for (int v = 0; v < 8; ++v) x[nt][v] += c6 * (acc[nt][v] + k[nt][v]);
    // trajectory t = s+1
#pragma unroll
    for (int nt = 0; nt < 2; ++nt)
#pragma unroll
      for (int v = 0; v < 8; ++v)
        out[((r0 + v + 8 * hi) * T + (s + 1)) * D + nt * 16 + nl] = x[nt][v];
  }
}

extern "C" void kernel_launch(void* const* d_in, const int* in_sizes, int n_in,
                              void* d_out, int out_size, void* d_ws, size_t ws_size,
                              hipStream_t stream) {
  (void)n_in; (void)out_size; (void)d_ws; (void)ws_size;
  const float* x0   = (const float*)d_in[0];
  const float* tsp  = (const float*)d_in[1];
  const float* W1   = (const float*)d_in[2];
  const float* b1   = (const float*)d_in[3];
  const float* W2   = (const float*)d_in[4];
  const float* b2   = (const float*)d_in[5];
  const float* W3   = (const float*)d_in[6];
  const float* b3   = (const float*)d_in[7];
  const float* Gw1  = (const float*)d_in[8];
  const float* Gb1  = (const float*)d_in[9];
  const float* Gw2  = (const float*)d_in[10];
  const float* Gb2  = (const float*)d_in[11];
  float* out = (float*)d_out;

  const int B = in_sizes[0] / D;                 // 16384
  const int nblocks = B / ROWS_PER_BLOCK;        // 256

  (void)hipFuncSetAttribute((const void*)AMEODE_90159953478503_kernel,
                            hipFuncAttributeMaxDynamicSharedMemorySize,
                            (int)SMEM_BYTES);

  AMEODE_90159953478503_kernel<<<nblocks, BLOCK, SMEM_BYTES, stream>>>(
      x0, tsp, W1, b1, W2, b2, W3, b3, Gw1, Gb1, Gw2, Gb2, out);
}
